// VarianceAdaptor_48129403518982
// MI455X (gfx1250) — matrix operand order repaired and checked by reading, not compiled, not run on hardware
//
#include <hip/hip_runtime.h>

typedef unsigned short ushort_t;
typedef unsigned int   uint_t;
typedef __attribute__((ext_vector_type(16))) __bf16 v16bf;
typedef __attribute__((ext_vector_type(8)))  float  v8f;

#define BB      32
#define TT      512
#define HH      256
#define BT      (BB*TT)
#define MAXLEN  4096

__device__ __forceinline__ ushort_t f2bf(float f){
  uint_t u = __float_as_uint(f);
  u += 0x7FFFu + ((u >> 16) & 1u);          // round-to-nearest-even
  return (ushort_t)(u >> 16);
}
__device__ __forceinline__ float bf2f(ushort_t h){
  return __uint_as_float(((uint_t)h) << 16);
}

// ---------------- f32 -> bf16 cast ----------------
__global__ void k_cast_bf16(const float* __restrict__ src, ushort_t* __restrict__ dst, int n){
  int i = blockIdx.x*256 + threadIdx.x;
  if (i < n) dst[i] = f2bf(src[i]);
}

// ------- conv weight (3,256,256) f32 -> chunk-major bf16 (24,[n=256],[k=32]) -------
__global__ void k_prep_w(const float* __restrict__ w, ushort_t* __restrict__ wc){
  int i = blockIdx.x*256 + threadIdx.x;        // 24*256*32 = 196608
  int j  = i & 31;
  int n  = (i >> 5) & 255;
  int c  = i >> 13;                            // chunk (8192 elems/chunk)
  int kw = c >> 3;
  int h  = ((c & 7) << 5) + j;
  wc[i] = f2bf(w[(kw*HH + h)*256 + n]);
}

// ---------------- fused conv1d(K=3) + bias + relu + LayerNorm ----------------
// A: (BT,256) bf16 activations, Wc: (24,256,32) bf16 weights (chunk major)
// block = 256 threads (8 waves); block tile = 32 rows x 256 cols; wave = 32x32
union FragU { uint4 u4[2]; v16bf v; };

__global__ void __launch_bounds__(256)
k_conv_ln(const ushort_t* __restrict__ A,
          const ushort_t* __restrict__ Wc,
          const float* __restrict__ bias,
          const float* __restrict__ gamma,
          const float* __restrict__ beta,
          ushort_t* __restrict__ O)
{
  __shared__ __align__(16) ushort_t lds_a[32*32];   // 2 KB  A chunk (m-major)
  __shared__ __align__(16) ushort_t lds_b[256*32];  // 16 KB B chunk (n-major)
  __shared__ float lds_o[32*256];                   // 32 KB post-relu tile
  __shared__ float red_s[256], red_q[256];
  __shared__ float stats[64];

  const int tid  = threadIdx.x;
  const int wave = tid >> 5;
  const int lane = tid & 31;
  const int hi   = lane >> 4;      // half-wave select
  const int ml   = lane & 15;

  const int rowBase = blockIdx.x * 32;       // 32-row tile stays inside one b (512%32==0)
  const int bidx = rowBase / TT;
  const int t0   = rowBase % TT;

  v8f acc00 = {}, acc01 = {}, acc10 = {}, acc11 = {};

  const unsigned ldsb_base = (unsigned)(unsigned long long)(&lds_b[0]); // LDS byte offset

  for (int c = 0; c < 24; ++c) {
    const int kw = c >> 3;
    const int dt = kw - 1;                   // conv pad = 1
    const int h0 = (c & 7) << 5;
    // ---- stage B (256 n x 32 k = 16KB) via CDNA5 async global->LDS copies ----
    {
      const unsigned long long gbase =
          (unsigned long long)(const char*)Wc + (unsigned long long)c * 16384ull;
      #pragma unroll
      for (int i = 0; i < 4; ++i) {
        int idx = tid + i*256;               // uint4 granule
        unsigned laddr = ldsb_base + (unsigned)(idx*16);
        unsigned long long gaddr = gbase + (unsigned long long)(idx*16);
        asm volatile("global_load_async_to_lds_b128 %0, %1, off"
                     :: "v"(laddr), "v"(gaddr) : "memory");
      }
    }
    // ---- stage A (32x32) with per-sequence im2col zero padding (predicated) ----
    {
      int m = tid >> 3, p = tid & 7;         // 8B per thread
      int t = t0 + m + dt;
      uint2 val; val.x = 0u; val.y = 0u;
      if (t >= 0 && t < TT)
        val = ((const uint2*)A)[((bidx*TT + t)*256 + h0)/4 + p];
      ((uint2*)lds_a)[m*8 + p] = val;
    }
    asm volatile("s_wait_asynccnt 0x0" ::: "memory");
    __syncthreads();
    // ---- gather WMMA fragments (ISA 16-bit A/B wave32 layouts) ----
    FragU af0, af1, bfr0, bfr1;
    {
      const int r0 = ml*32;                  // rows 0..15 of tile
      af0.u4[0] = *(const uint4*)&lds_a[r0 + hi*8];
      af0.u4[1] = *(const uint4*)&lds_a[r0 + 16 + hi*8];
      const int r1 = (16 + ml)*32;           // rows 16..31 of tile
      af1.u4[0] = *(const uint4*)&lds_a[r1 + hi*8];
      af1.u4[1] = *(const uint4*)&lds_a[r1 + 16 + hi*8];
    }
    {
      int n0 = wave*32 + ml;                 // B: lane ml -> col, K = hi*8 + 0..7, 16 + hi*8 + 0..7
      bfr0.u4[0] = *(const uint4*)&lds_b[n0*32 + hi*8];
      bfr0.u4[1] = *(const uint4*)&lds_b[n0*32 + 16 + hi*8];
      int n1 = n0 + 16;
      bfr1.u4[0] = *(const uint4*)&lds_b[n1*32 + hi*8];
      bfr1.u4[1] = *(const uint4*)&lds_b[n1*32 + 16 + hi*8];
    }
    acc00 = __builtin_amdgcn_wmma_f32_16x16x32_bf16(false, af0.v, false, bfr0.v,
                                                    (short)0, acc00, false, false);
    acc01 = __builtin_amdgcn_wmma_f32_16x16x32_bf16(false, af0.v, false, bfr1.v,
                                                    (short)0, acc01, false, false);
    acc10 = __builtin_amdgcn_wmma_f32_16x16x32_bf16(false, af1.v, false, bfr0.v,
                                                    (short)0, acc10, false, false);
    acc11 = __builtin_amdgcn_wmma_f32_16x16x32_bf16(false, af1.v, false, bfr1.v,
                                                    (short)0, acc11, false, false);
    __syncthreads();
  }

  // ---- epilogue: bias + relu -> LDS (C layout: vgpr r, row = r + 8*hi, col = ml) ----
  {
    int n0 = wave*32 + ml, n1 = n0 + 16;
    float bb0 = bias[n0], bb1 = bias[n1];
    #pragma unroll
    for (int r = 0; r < 8; ++r) {
      int row0 = r + hi*8;
      int row1 = 16 + row0;
      float v;
      v = acc00[r] + bb0; lds_o[row0*256 + n0] = v > 0.f ? v : 0.f;
      v = acc01[r] + bb1; lds_o[row0*256 + n1] = v > 0.f ? v : 0.f;
      v = acc10[r] + bb0; lds_o[row1*256 + n0] = v > 0.f ? v : 0.f;
      v = acc11[r] + bb1; lds_o[row1*256 + n1] = v > 0.f ? v : 0.f;
    }
  }
  __syncthreads();
  // ---- LayerNorm over 256 features per row (32 rows, 8 threads/row) ----
  {
    int r = tid >> 3, s = tid & 7;
    float sm = 0.f, sq = 0.f;
    #pragma unroll
    for (int i = 0; i < 32; ++i) {
      float v = lds_o[r*256 + s*32 + i];
      sm += v; sq += v*v;
    }
    red_s[r*8+s] = sm; red_q[r*8+s] = sq;
  }
  __syncthreads();
  if (tid < 32) {
    float sm = 0.f, sq = 0.f;
    #pragma unroll
    for (int i = 0; i < 8; ++i) { sm += red_s[tid*8+i]; sq += red_q[tid*8+i]; }
    float mean = sm * (1.f/256.f);
    float var  = sq * (1.f/256.f) - mean*mean;
    stats[tid*2]   = mean;
    stats[tid*2+1] = rsqrtf(var + 1e-5f);
  }
  __syncthreads();
  {
    int r = tid >> 3, s = tid & 7;
    float mean = stats[r*2], rstd = stats[r*2+1];
    int grow = rowBase + r;
    #pragma unroll
    for (int i = 0; i < 32; ++i) {
      int col = s*32 + i;
      float v = (lds_o[r*256 + col] - mean) * rstd * gamma[col] + beta[col];
      O[grow*256 + col] = f2bf(v);
    }
  }
}

// ---------------- final F->1 linear + mask (wave per row) ----------------
__global__ void __launch_bounds__(256)
k_linear(const ushort_t* __restrict__ Hb, const float* __restrict__ lw,
         const float* __restrict__ lb, const unsigned char* __restrict__ mask,
         float* __restrict__ out)
{
  int wave = threadIdx.x >> 5, lane = threadIdx.x & 31;
  int row = blockIdx.x*8 + wave;
  float acc = 0.f;
  #pragma unroll
  for (int i = 0; i < 8; ++i) {
    int c = lane + i*32;
    acc += bf2f(Hb[row*256 + c]) * lw[c];
  }
  #pragma unroll
  for (int off = 16; off; off >>= 1) acc += __shfl_xor(acc, off, 32);
  if (lane == 0) out[row] = mask[row] ? 0.f : (acc + lb[0]);
}

// ---------------- x += emb[searchsorted(bins, tgt, 'left')] ----------------
__global__ void __launch_bounds__(256)
k_emb_add(const float* __restrict__ xsrc, const float* __restrict__ tgt,
          const float* __restrict__ bins, const float* __restrict__ emb,
          float* __restrict__ xdst, ushort_t* __restrict__ xbf)
{
  int row = blockIdx.x;
  float t = tgt[row];
  int lo = 0, hiN = 255;                     // 255 bin edges, idx in [0,255]
  while (lo < hiN) { int mid = (lo + hiN) >> 1; if (bins[mid] < t) lo = mid + 1; else hiN = mid; }
  int col = threadIdx.x;
  float v = xsrc[row*256 + col] + emb[lo*256 + col];
  xdst[row*256 + col] = v;
  xbf[row*256 + col]  = f2bf(v);
}

// ---------------- per-sequence cumsum + duration passthrough + mel_len ----------------
__global__ void k_cumsum(const int* __restrict__ dur, int* __restrict__ cum,
                         float* __restrict__ dur_out, float* __restrict__ mel_len)
{
  int b = threadIdx.x;
  if (b >= BB) return;
  int acc = 0;
  for (int t = 0; t < TT; ++t) {
    int d = dur[b*TT + t];
    acc += d;
    cum[b*TT + t]     = acc;
    dur_out[b*TT + t] = (float)d;
  }
  mel_len[b] = (float)acc;
}

// ---------------- length regulate: gather + mask ----------------
__global__ void __launch_bounds__(256)
k_regulate(const float* __restrict__ x, const int* __restrict__ cum,
           float* __restrict__ out, float* __restrict__ mel_mask)
{
  __shared__ int scum[TT];
  int b = blockIdx.y;
  for (int i = threadIdx.x; i < TT; i += 256) scum[i] = cum[b*TT + i];
  __syncthreads();
  int grp = threadIdx.x >> 5, lane = threadIdx.x & 31;
  int pos = blockIdx.x*8 + grp;
  int mlen = scum[TT-1];
  int lo = 0, hi = TT;                        // searchsorted 'right'
  while (lo < hi) { int mid = (lo+hi)>>1; if (scum[mid] <= pos) lo = mid+1; else hi = mid; }
  int idx = lo < TT-1 ? lo : TT-1;
  bool valid = pos < mlen;
  const float4* srow = (const float4*)(x + ((size_t)b*TT + idx)*256);
  float4*       drow = (float4*)(out + ((size_t)b*MAXLEN + pos)*256);
  float4 z = {0.f,0.f,0.f,0.f};
  #pragma unroll
  for (int i = 0; i < 2; ++i) {
    int c4 = lane + i*32;
    drow[c4] = valid ? srow[c4] : z;
  }
  if (lane == 0) mel_mask[(size_t)b*MAXLEN + pos] = valid ? 1.f : 0.f;
}

extern "C" void kernel_launch(void* const* d_in, const int* in_sizes, int n_in,
                              void* d_out, int out_size, void* d_ws, size_t ws_size,
                              hipStream_t stream)
{
  const float* x               = (const float*)d_in[0];
  const unsigned char* src_msk = (const unsigned char*)d_in[1];
  const float* pitch_t         = (const float*)d_in[2];
  const float* energy_t        = (const float*)d_in[3];
  const int*   dur_t           = (const int*)d_in[4];
  // d_in[5] = max_len (constant 4096)
  const float* pitch_emb  = (const float*)d_in[36];
  const float* energy_emb = (const float*)d_in[37];
  const float* pitch_bins = (const float*)d_in[38];
  const float* energy_bins= (const float*)d_in[39];

  // workspace layout
  char* ws = (char*)d_ws;
  size_t off = 0;
  auto alloc = [&](size_t bytes){ void* p = ws + off; off = (off + bytes + 255) & ~(size_t)255; return p; };
  ushort_t* xbf  = (ushort_t*)alloc((size_t)BT*256*2);
  ushort_t* h1   = (ushort_t*)alloc((size_t)BT*256*2);
  ushort_t* h2   = (ushort_t*)alloc((size_t)BT*256*2);
  float*    xacc = (float*)   alloc((size_t)BT*256*4);
  ushort_t* wt1  = (ushort_t*)alloc((size_t)24*256*32*2);
  ushort_t* wt2  = (ushort_t*)alloc((size_t)24*256*32*2);
  int*      cum  = (int*)     alloc((size_t)BB*TT*4);

  // output slices (reference tuple order, flattened)
  float* out      = (float*)d_out;
  float* o_mel    = out;                              // (32,4096,256)
  float* o_pitch  = out + (size_t)BB*MAXLEN*256;      // (32,512)
  float* o_energy = o_pitch + BT;
  float* o_logdur = o_energy + BT;
  float* o_durf   = o_logdur + BT;                    // duration passthrough
  float* o_mellen = o_durf + BT;                      // (32,)
  float* o_mask   = o_mellen + BB;                    // (32,4096)

  k_cast_bf16<<<(BT*256)/256, 256, 0, stream>>>(x, xbf, BT*256);

  auto run_predictor = [&](int base, const ushort_t* in_bf, float* pred_out){
    const float* b1  = (const float*)d_in[base+1];
    const float* g1  = (const float*)d_in[base+2];
    const float* be1 = (const float*)d_in[base+3];
    const float* b2  = (const float*)d_in[base+5];
    const float* g2  = (const float*)d_in[base+6];
    const float* be2 = (const float*)d_in[base+7];
    const float* lw  = (const float*)d_in[base+8];
    const float* lb  = (const float*)d_in[base+9];
    k_prep_w<<<768, 256, 0, stream>>>((const float*)d_in[base+0], wt1);
    k_prep_w<<<768, 256, 0, stream>>>((const float*)d_in[base+4], wt2);
    k_conv_ln<<<BT/32, 256, 0, stream>>>(in_bf, wt1, b1, g1, be1, h1);
    k_conv_ln<<<BT/32, 256, 0, stream>>>(h1,   wt2, b2, g2, be2, h2);
    k_linear<<<BT/8, 256, 0, stream>>>(h2, lw, lb, src_msk, pred_out);
  };

  run_predictor(6,  xbf, o_logdur);                   // dur predictor (uses original x)
  run_predictor(16, xbf, o_pitch);                    // pitch predictor (original x)
  k_emb_add<<<BT, 256, 0, stream>>>(x, pitch_t, pitch_bins, pitch_emb, xacc, xbf);
  run_predictor(26, xbf, o_energy);                   // energy predictor (x + pitch emb)
  k_emb_add<<<BT, 256, 0, stream>>>(xacc, energy_t, energy_bins, energy_emb, xacc, xbf);
  k_cumsum<<<1, 32, 0, stream>>>(dur_t, cum, o_durf, o_mellen);
  dim3 rg(MAXLEN/8, BB);
  k_regulate<<<rg, 256, 0, stream>>>(xacc, cum, o_mel, o_mask);
}
